// SchNet_GNN_Interact_before_41618233099046
// MI455X (gfx1250) — matrix operand order repaired and checked by reading, not compiled, not run on hardware
//
#include <hip/hip_runtime.h>

// ---------------- problem constants (from reference) ----------------
#define NNODES 16384
#define NGRAPH 512
#define NE3    262144
#define NE2    65536
#define HIDDIM 128
#define NFILT  128
#define NGAUSS 50
#define NODEF  39
#define BONDF  10
#define FPSZ   2048
#define NTASK  12
#define NLAYER 6
#define RCUT   10.0f
#define PI_F   3.14159265358979f

typedef __attribute__((ext_vector_type(16))) _Float16 v16h;
typedef __attribute__((ext_vector_type(8)))  _Float16 v8h;
typedef __attribute__((ext_vector_type(8)))  float    v8f;

// shifted softplus: softplus(x) - ln(2)
__device__ __forceinline__ float dssp(float x) {
  float sp = (x > 20.0f) ? x : log1pf(expf(x));
  return sp - 0.6931471805599453f;
}

__device__ __forceinline__ v16h combine16(v8h lo, v8h hi) {
  v16h a;
#pragma unroll
  for (int i = 0; i < 8; ++i) { a[i] = lo[i]; a[8 + i] = hi[i]; }
  return a;
}

// ---------------- WMMA fragment loaders (all unconditional, vectorized) --
// A layout (CDNA5 ISA): lane holds row M=l&15; halves are two contiguous
// K-runs of 8: [k0 + hs*8 .. +7] and [k0+16+hs*8 .. +7].
__device__ __forceinline__ v16h load_a(const float* __restrict__ A, int lda,
                                       int m0, int k0) {
  const int lane = threadIdx.x & 31;
  const float* p = A + (long)(m0 + (lane & 15)) * lda + k0 + ((lane >> 4) << 3);
  const float4 q0 = *(const float4*)(p);
  const float4 q1 = *(const float4*)(p + 4);
  const float4 q2 = *(const float4*)(p + 16);
  const float4 q3 = *(const float4*)(p + 20);
  v16h a;
  a[0]=(_Float16)q0.x;  a[1]=(_Float16)q0.y;  a[2]=(_Float16)q0.z;  a[3]=(_Float16)q0.w;
  a[4]=(_Float16)q1.x;  a[5]=(_Float16)q1.y;  a[6]=(_Float16)q1.z;  a[7]=(_Float16)q1.w;
  a[8]=(_Float16)q2.x;  a[9]=(_Float16)q2.y;  a[10]=(_Float16)q2.z; a[11]=(_Float16)q2.w;
  a[12]=(_Float16)q3.x; a[13]=(_Float16)q3.y; a[14]=(_Float16)q3.z; a[15]=(_Float16)q3.w;
  return a;
}
__device__ __forceinline__ v16h load_a(const _Float16* __restrict__ A, int lda,
                                       int m0, int k0) {
  const int lane = threadIdx.x & 31;
  const _Float16* p = A + (long)(m0 + (lane & 15)) * lda + k0 + ((lane >> 4) << 3);
  return combine16(*(const v8h*)p, *(const v8h*)(p + 16));
}
// B pre-swizzled: tile (kt,nt) is 512 halves laid out [lane][16 halves].
__device__ __forceinline__ v16h load_b_sw(const _Float16* __restrict__ Bsw,
                                          int kt, int nt, int tilesN) {
  const int lane = threadIdx.x & 31;
  return *(const v16h*)(Bsw + (((long)kt * tilesN + nt) << 9) + (lane << 4));
}

// ---------------- weight pre-swizzle into B-fragment layout --------------
__global__ void swizzle_b_kernel(const float* __restrict__ W, int K, int N,
                                 int Kpad, int Npad, _Float16* __restrict__ out) {
  const int lane  = threadIdx.x & 31;
  const int wave  = blockIdx.x * (blockDim.x >> 5) + (threadIdx.x >> 5);
  const int tilesN = Npad >> 4;
  const int kt = wave / tilesN, nt = wave % tilesN;
  if (kt >= (Kpad >> 5)) return;
  const int c = (nt << 4) + (lane & 15);
  const int kbase = (kt << 5) + ((lane >> 4) << 3);
  _Float16* o = out + (((long)kt * tilesN + nt) << 9) + (lane << 4);
  if (c < N) {
#pragma unroll
    for (int h = 0; h < 16; ++h) {
      const int k = kbase + (h & 7) + ((h >> 3) << 4);
      o[h] = (k < K) ? (_Float16)W[(long)k * N + c] : (_Float16)0.0f;
    }
  } else {
#pragma unroll
    for (int h = 0; h < 16; ++h) o[h] = (_Float16)0.0f;
  }
}

// ---------------- generic f16 pad-copy (zero-fills K..Kpad) --------------
__global__ void pad_f16_kernel(const float* __restrict__ src, _Float16* __restrict__ dst,
                               long M, int K, int Kpad) {
  const long i = (long)blockIdx.x * blockDim.x + threadIdx.x;
  if (i >= M * Kpad) return;
  const long m = i / Kpad; const int k = (int)(i - m * Kpad);
  dst[i] = (k < K) ? (_Float16)src[m * K + k] : (_Float16)0.0f;
}

// ---------------- generic GEMM: C = act(A@Bsw + bias) [+C if beta] -------
// A: f32 or f16 row-major, lda % 16 == 0, K zero-padded to Kpad (mult 32).
template <typename AT>
__global__ void __launch_bounds__(256)
gemm_sw_kernel(const AT* __restrict__ A, const _Float16* __restrict__ Bsw,
               const float* __restrict__ bias, float* __restrict__ C,
               int M, int N, int Kpad, int lda, int ldc, int act, int beta) {
  const int wave   = blockIdx.x * (blockDim.x >> 5) + (threadIdx.x >> 5);
  const int tilesN = (N + 15) >> 4;
  const int m0 = (wave / tilesN) << 4;
  const int tn =  wave % tilesN;
  if (m0 >= M) return;                        // whole-wave uniform exit
  v8f acc = {};
  const int ktiles = Kpad >> 5;
  for (int kt = 0; kt < ktiles; ++kt) {
    if (kt + 1 < ktiles)
      __builtin_prefetch((const void*)(Bsw + (((long)(kt + 1) * tilesN + tn) << 9)), 0, 3);
    v16h a = load_a(A, lda, m0, kt << 5);
    v16h b = load_b_sw(Bsw, kt, tn, tilesN);
    acc = __builtin_amdgcn_wmma_f32_16x16x32_f16(false, a, false, b,
                                                 (short)0, acc, false, false);
  }
  const int lane = threadIdx.x & 31;
  const int c = (tn << 4) + (lane & 15);
  if (c >= N) return;                         // only the NTASK=12 tail
  const float bv = bias ? bias[c] : 0.0f;
#pragma unroll
  for (int r = 0; r < 8; ++r) {
    const long mr = m0 + r + ((lane >> 4) << 3);
    float val = acc[r] + bv;
    if (act == 1) val = fmaxf(val, 0.0f);
    else if (act == 2) val = dssp(val);
    if (beta) val += C[mr * ldc + c];
    C[mr * ldc + c] = val;
  }
}

// ---------------- edge geometry: f16 padded smearing + cos cutoff --------
__global__ void edge_geom_kernel(const float* __restrict__ pos,
                                 const int* __restrict__ row,
                                 const int* __restrict__ col,
                                 _Float16* __restrict__ demb,   // [E3,64] padded
                                 float* __restrict__ env) {
  const int e = blockIdx.x * blockDim.x + threadIdx.x;
  if (e >= NE3) return;
  const int r = row[e], c = col[e];
  const float dx = pos[3*r]   - pos[3*c];
  const float dy = pos[3*r+1] - pos[3*c+1];
  const float dz = pos[3*r+2] - pos[3*c+2];
  const float dist = sqrtf(dx*dx + dy*dy + dz*dz);
  const float step  = RCUT / (float)(NGAUSS - 1);
  const float coeff = -0.5f / (step * step);
  const long base = (long)e * 64;
#pragma unroll
  for (int g = 0; g < NGAUSS; ++g) {
    const float d = dist - (float)g * step;
    demb[base + g] = (_Float16)expf(coeff * d * d);
  }
#pragma unroll
  for (int g = NGAUSS; g < 64; ++g) demb[base + g] = (_Float16)0.0f;
  env[e] = 0.5f * (cosf(dist * PI_F / RCUT) + 1.0f);
}

// ---------------- init: cat[:,128:256] = emb[z] --------------------------
__global__ void init_cat_kernel(const int* __restrict__ z,
                                const float* __restrict__ emb,
                                float* __restrict__ cat) {
  const long i = (long)blockIdx.x * blockDim.x + threadIdx.x;
  if (i >= (long)NNODES * HIDDIM) return;
  const int n = (int)(i >> 7), j = (int)(i & 127);
  cat[(long)n * 256 + 128 + j] = emb[(long)z[n] * HIDDIM + j];
}

// ---------------- fused SchNet edge pipeline (hot loop) ------------------
// Per wave: 16 edges. H = ssp(demb@W1+b1) in LDS (f16), F = (H@W2+b2)*env,
// e = F * t[row], atomicAdd into agg[col].
__global__ void __launch_bounds__(256)
schnet_edge_kernel(const _Float16* __restrict__ demb, const float* __restrict__ env,
                   const _Float16* __restrict__ w1sw, const float* __restrict__ b1,
                   const _Float16* __restrict__ w2sw, const float* __restrict__ b2,
                   const float* __restrict__ t,  const int* __restrict__ row,
                   const int* __restrict__ col,  float* __restrict__ agg) {
  __shared__ _Float16 lh[8][16][NFILT];            // 32 KB
  const int w    = threadIdx.x >> 5;
  const int lane = threadIdx.x & 31;
  const int hs   = lane >> 4;
  const int m0   = (blockIdx.x * 8 + w) << 4;      // exact grid: NE3/16 waves
  v8f zero8 = {};
  // ---- stage 1: [16x128] = demb[16x64pad] @ W1sw
  v8f acc[8];
#pragma unroll
  for (int nt = 0; nt < 8; ++nt) acc[nt] = zero8;
#pragma unroll
  for (int kt = 0; kt < 2; ++kt) {
    v16h a = load_a(demb, 64, m0, kt << 5);
#pragma unroll
    for (int nt = 0; nt < 8; ++nt)
      acc[nt] = __builtin_amdgcn_wmma_f32_16x16x32_f16(false, a, false,
                  load_b_sw(w1sw, kt, nt, 8), (short)0, acc[nt], false, false);
  }
  // bias + ssp -> LDS (row-major 16x128 f16)
#pragma unroll
  for (int nt = 0; nt < 8; ++nt) {
    const int cc = (nt << 4) + (lane & 15);
    const float bv = b1[cc];
#pragma unroll
    for (int r = 0; r < 8; ++r)
      lh[w][r + (hs << 3)][cc] = (_Float16)dssp(acc[nt][r] + bv);
  }
  __syncthreads();
  // ---- stage 2: [16x128] = H[16x128] @ W2sw
  v8f facc[8];
#pragma unroll
  for (int nt = 0; nt < 8; ++nt) facc[nt] = zero8;
#pragma unroll
  for (int kt = 0; kt < 4; ++kt) {
    const int k0 = kt << 5;
    const _Float16* lp = &lh[w][lane & 15][k0 + (hs << 3)];
    v16h a = combine16(*(const v8h*)lp, *(const v8h*)(lp + 16));
#pragma unroll
    for (int nt = 0; nt < 8; ++nt)
      facc[nt] = __builtin_amdgcn_wmma_f32_16x16x32_f16(false, a, false,
                   load_b_sw(w2sw, kt, nt, 8), (short)0, facc[nt], false, false);
  }
  // ---- epilogue: cutoff envelope, gather t[row], scatter-add agg[col]
#pragma unroll
  for (int r = 0; r < 8; ++r) {
    const int  me   = m0 + r + (hs << 3);
    const float cen = env[me];
    const long trow = (long)row[me] * NFILT;
    const long crow = (long)col[me] * NFILT;
#pragma unroll
    for (int nt = 0; nt < 8; ++nt) {
      const int cc = (nt << 4) + (lane & 15);
      const float W = (facc[nt][r] + b2[cc]) * cen;
      atomicAdd(&agg[crow + cc], W * t[trow + cc]);
    }
  }
}

// ---------------- fused GINE edge kernel ---------------------------------
__global__ void __launch_bounds__(256)
gine_edge_kernel(const _Float16* __restrict__ epad, const _Float16* __restrict__ wlsw,
                 const float* __restrict__ bl, const float* __restrict__ hid,
                 const int* __restrict__ src, const int* __restrict__ dst,
                 float* __restrict__ agg2) {
  const int lane = threadIdx.x & 31;
  const int hs   = lane >> 4;
  const int m0   = (blockIdx.x * (blockDim.x >> 5) + (threadIdx.x >> 5)) << 4;
  v8f zero8 = {};
  v8f acc[8];
#pragma unroll
  for (int nt = 0; nt < 8; ++nt) acc[nt] = zero8;
  v16h a = load_a(epad, 32, m0, 0);      // K=10 zero-padded to 32: single step
#pragma unroll
  for (int nt = 0; nt < 8; ++nt)
    acc[nt] = __builtin_amdgcn_wmma_f32_16x16x32_f16(false, a, false,
                load_b_sw(wlsw, 0, nt, 8), (short)0, acc[nt], false, false);
#pragma unroll
  for (int r = 0; r < 8; ++r) {
    const int  me   = m0 + r + (hs << 3);
    const long srow = (long)src[me] * HIDDIM;
    const long drow = (long)dst[me] * HIDDIM;
#pragma unroll
    for (int nt = 0; nt < 8; ++nt) {
      const int cc = (nt << 4) + (lane & 15);
      const float msg = fmaxf(acc[nt][r] + bl[cc] + hid[srow + cc], 0.0f);
      atomicAdd(&agg2[drow + cc], msg);
    }
  }
}

// ---------------- elementwise / pooling kernels --------------------------
__global__ void zero_kernel(float* __restrict__ p, long n) {
  const long i = (long)blockIdx.x * blockDim.x + threadIdx.x;
  if (i < n) p[i] = 0.0f;
}
__global__ void add_kernel(const float* __restrict__ a, const float* __restrict__ b,
                           float* __restrict__ c, long n) {
  const long i = (long)blockIdx.x * blockDim.x + threadIdx.x;
  if (i < n) c[i] = a[i] + b[i];
}
__global__ void pool_kernel(const float* __restrict__ x, const int* __restrict__ batch,
                            float* __restrict__ out) {
  const long i = (long)blockIdx.x * blockDim.x + threadIdx.x;
  if (i >= (long)NNODES * HIDDIM) return;
  const int n = (int)(i >> 7), j = (int)(i & 127);
  atomicAdd(&out[(long)batch[n] * HIDDIM + j], x[i]);
}
__global__ void cat3_kernel(const float* __restrict__ g, const float* __restrict__ f,
                            const float* __restrict__ s, float* __restrict__ cat) {
  const int i = blockIdx.x * blockDim.x + threadIdx.x;
  if (i >= NGRAPH * 3 * HIDDIM) return;
  const int r = i / (3 * HIDDIM), c = i % (3 * HIDDIM);
  float v;
  if (c < HIDDIM)          v = g[r * HIDDIM + c];
  else if (c < 2 * HIDDIM) v = f[r * HIDDIM + c - HIDDIM];
  else                     v = s[r * HIDDIM + c - 2 * HIDDIM];
  cat[i] = v;
}

// ---------------- host-side helpers --------------------------------------
static void gemm_f32(hipStream_t s, const float* A, const _Float16* Bsw,
                     const float* bias, float* C, int M, int N, int Kpad,
                     int lda, int ldc, int act, int beta) {
  const int tiles = ((M + 15) / 16) * ((N + 15) / 16);
  gemm_sw_kernel<float><<<(tiles + 7) / 8, 256, 0, s>>>(A, Bsw, bias, C, M, N, Kpad,
                                                        lda, ldc, act, beta);
}
static void gemm_f16(hipStream_t s, const _Float16* A, const _Float16* Bsw,
                     const float* bias, float* C, int M, int N, int Kpad,
                     int lda, int ldc, int act, int beta) {
  const int tiles = ((M + 15) / 16) * ((N + 15) / 16);
  gemm_sw_kernel<_Float16><<<(tiles + 7) / 8, 256, 0, s>>>(A, Bsw, bias, C, M, N, Kpad,
                                                           lda, ldc, act, beta);
}
static void zero(hipStream_t s, float* p, long n) {
  zero_kernel<<<(int)((n + 255) / 256), 256, 0, s>>>(p, n);
}

// input order: 30 params (dict insertion order) then z,pos,x,ei3,ei2,eattr2,batch,fp
enum {
  P_EMB = 0, P_NODE_W, P_NODE_B, P_AGGR_W, P_AGGR_B,
  P_UE_LIN_W, P_UE_W1, P_UE_B1, P_UE_W2, P_UE_B2,
  P_UV1_W, P_UV1_B, P_UV2_W, P_UV2_B,
  P_EL_W, P_EL_B,
  P_UU_W1, P_UU_B1, P_UU_W2, P_UU_B2,
  P_NN_W, P_NN_B, P_FPN_W1, P_FPN_B1, P_FPN_W2, P_FPN_B2,
  P_HID_W, P_HID_B, P_OUT_W, P_OUT_B,
  I_Z, I_POS, I_X, I_EI3, I_EI2, I_EATTR2, I_BATCH, I_FP
};

extern "C" void kernel_launch(void* const* d_in, const int* in_sizes, int n_in,
                              void* d_out, int out_size, void* d_ws, size_t ws_size,
                              hipStream_t stream) {
  const float* emb     = (const float*)d_in[P_EMB];
  const float* node_w  = (const float*)d_in[P_NODE_W];
  const float* node_b  = (const float*)d_in[P_NODE_B];
  const float* aggr_w  = (const float*)d_in[P_AGGR_W];
  const float* aggr_b  = (const float*)d_in[P_AGGR_B];
  const float* uelin_w = (const float*)d_in[P_UE_LIN_W];
  const float* ue_w1   = (const float*)d_in[P_UE_W1];
  const float* ue_b1   = (const float*)d_in[P_UE_B1];
  const float* ue_w2   = (const float*)d_in[P_UE_W2];
  const float* ue_b2   = (const float*)d_in[P_UE_B2];
  const float* uv1_w   = (const float*)d_in[P_UV1_W];
  const float* uv1_b   = (const float*)d_in[P_UV1_B];
  const float* uv2_w   = (const float*)d_in[P_UV2_W];
  const float* uv2_b   = (const float*)d_in[P_UV2_B];
  const float* el_w    = (const float*)d_in[P_EL_W];
  const float* el_b    = (const float*)d_in[P_EL_B];
  const float* uu_w1   = (const float*)d_in[P_UU_W1];
  const float* uu_b1   = (const float*)d_in[P_UU_B1];
  const float* uu_w2   = (const float*)d_in[P_UU_W2];
  const float* uu_b2   = (const float*)d_in[P_UU_B2];
  const float* nn_w    = (const float*)d_in[P_NN_W];
  const float* nn_b    = (const float*)d_in[P_NN_B];
  const float* fpn_w1  = (const float*)d_in[P_FPN_W1];
  const float* fpn_b1  = (const float*)d_in[P_FPN_B1];
  const float* fpn_w2  = (const float*)d_in[P_FPN_W2];
  const float* fpn_b2  = (const float*)d_in[P_FPN_B2];
  const float* hid_w   = (const float*)d_in[P_HID_W];
  const float* hid_b   = (const float*)d_in[P_HID_B];
  const float* out_w   = (const float*)d_in[P_OUT_W];
  const float* out_b   = (const float*)d_in[P_OUT_B];
  const int*   z       = (const int*)d_in[I_Z];
  const float* pos     = (const float*)d_in[I_POS];
  const float* x       = (const float*)d_in[I_X];
  const int*   e3row   = (const int*)d_in[I_EI3];
  const int*   e3col   = e3row + NE3;
  const int*   e2src   = (const int*)d_in[I_EI2];
  const int*   e2dst   = e2src + NE2;
  const float* eattr2  = (const float*)d_in[I_EATTR2];
  const int*   batch   = (const int*)d_in[I_BATCH];
  const float* fp      = (const float*)d_in[I_FP];

  // output layout (tuple flattened): out[G,12], gnn_out[G,128], sch_out[G,128]
  float* out_p = (float*)d_out;
  float* gnn_p = out_p + (long)NGRAPH * NTASK;
  float* sch_p = gnn_p + (long)NGRAPH * HIDDIM;

  // ---- workspace carve-up ----
  char* wsb = (char*)d_ws;
  size_t off = 0;
  auto allocf = [&](long n) { float* p = (float*)(wsb + off);
                              off = (off + (size_t)n * 4 + 255) & ~(size_t)255; return p; };
  auto alloch = [&](long n) { _Float16* p = (_Float16*)(wsb + off);
                              off = (off + (size_t)n * 2 + 255) & ~(size_t)255; return p; };

  _Float16* demb16 = alloch((long)NE3 * 64);
  float*    env    = allocf(NE3);
  _Float16* xpad   = alloch((long)NNODES * 64);
  _Float16* epad   = alloch((long)NE2 * 32);
  float*    cat2   = allocf((long)NNODES * 256);
  float*    vbuf   = allocf((long)NNODES * HIDDIM);
  float*    hbuf   = allocf((long)NNODES * HIDDIM);
  float*    tbuf   = allocf((long)NNODES * HIDDIM);
  float*    agg    = allocf((long)NNODES * HIDDIM);
  float*    nb1    = allocf((long)NNODES * HIDDIM);
  float*    nb2    = allocf((long)NNODES * HIDDIM);
  float*    gt1    = allocf((long)NGRAPH * HIDDIM);
  float*    fcn    = allocf((long)NGRAPH * HIDDIM);
  float*    cat3   = allocf((long)NGRAPH * 3 * HIDDIM);
  float*    gt2    = allocf((long)NGRAPH * HIDDIM);

  // swizzle helper: W[K,N] (row-major f32) -> f16 B-fragment tiles
  auto swz = [&](const float* W, int K, int N, int Kpad, int Npad) -> _Float16* {
    _Float16* dst = alloch((long)(Kpad >> 5) * (Npad >> 4) * 512);
    const int tiles = (Kpad >> 5) * (Npad >> 4);
    swizzle_b_kernel<<<(tiles + 7) / 8, 256, 0, stream>>>(W, K, N, Kpad, Npad, dst);
    return dst;
  };

  _Float16* node_sw = swz(node_w, NODEF, HIDDIM, 64, HIDDIM);
  _Float16* aggr_sw = swz(aggr_w, 256, HIDDIM, 256, HIDDIM);
  _Float16 *uelin_sw[NLAYER], *w1_sw[NLAYER], *w2_sw[NLAYER],
           *uv1_sw[NLAYER], *uv2_sw[NLAYER], *el_sw[NLAYER];
  for (int i = 0; i < NLAYER; ++i) {
    uelin_sw[i] = swz(uelin_w + (long)i * HIDDIM * NFILT, HIDDIM, NFILT, HIDDIM, NFILT);
    w1_sw[i]    = swz(ue_w1 + (long)i * NGAUSS * NFILT, NGAUSS, NFILT, 64, NFILT);
    w2_sw[i]    = swz(ue_w2 + (long)i * NFILT * NFILT, NFILT, NFILT, NFILT, NFILT);
    uv1_sw[i]   = swz(uv1_w + (long)i * NFILT * HIDDIM, NFILT, HIDDIM, NFILT, HIDDIM);
    uv2_sw[i]   = swz(uv2_w + (long)i * HIDDIM * HIDDIM, HIDDIM, HIDDIM, HIDDIM, HIDDIM);
    el_sw[i]    = swz(el_w + (long)i * BONDF * HIDDIM, BONDF, HIDDIM, 32, HIDDIM);
  }
  _Float16* nn_sw   = swz(nn_w, HIDDIM, HIDDIM, HIDDIM, HIDDIM);
  _Float16* uu1_sw  = swz(uu_w1, HIDDIM, HIDDIM / 2, HIDDIM, HIDDIM / 2);
  _Float16* uu2_sw  = swz(uu_w2, HIDDIM / 2, HIDDIM, HIDDIM / 2, HIDDIM);
  _Float16* fpn1_sw = swz(fpn_w1, FPSZ, HIDDIM, FPSZ, HIDDIM);
  _Float16* fpn2_sw = swz(fpn_w2, HIDDIM, HIDDIM, HIDDIM, HIDDIM);
  _Float16* hidw_sw = swz(hid_w, 3 * HIDDIM, HIDDIM, 3 * HIDDIM, HIDDIM);
  _Float16* outw_sw = swz(out_w, HIDDIM, NTASK, HIDDIM, 16);
  (void)ws_size; (void)in_sizes; (void)n_in; (void)out_size;

  const long NH = (long)NNODES * HIDDIM;

  // 1) edge geometry + padded A-side activations
  edge_geom_kernel<<<NE3 / 256, 256, 0, stream>>>(pos, e3row, e3col, demb16, env);
  pad_f16_kernel<<<(int)(((long)NNODES * 64 + 255) / 256), 256, 0, stream>>>(
      x, xpad, NNODES, NODEF, 64);
  pad_f16_kernel<<<(int)(((long)NE2 * 32 + 255) / 256), 256, 0, stream>>>(
      eattr2, epad, NE2, BONDF, 32);

  // 2) node init: cat = [x@W+b, emb[z]]; v = cat@aggr_w + b; hid = v
  gemm_f16(stream, xpad, node_sw, node_b, cat2, NNODES, HIDDIM, 64, 64, 256, 0, 0);
  init_cat_kernel<<<(int)(NH / 256), 256, 0, stream>>>(z, emb, cat2);
  gemm_f32(stream, cat2, aggr_sw, aggr_b, vbuf, NNODES, HIDDIM, 256, 256, HIDDIM, 0, 0);
  hipMemcpyAsync(hbuf, vbuf, NH * sizeof(float), hipMemcpyDeviceToDevice, stream);

  // 3) SchNet interaction blocks
  for (int i = 0; i < NLAYER; ++i) {
    gemm_f32(stream, vbuf, uelin_sw[i], nullptr, tbuf, NNODES, NFILT, HIDDIM,
             HIDDIM, NFILT, 0, 0);
    zero(stream, agg, NH);
    schnet_edge_kernel<<<NE3 / (16 * 8), 256, 0, stream>>>(
        demb16, env, w1_sw[i], ue_b1 + (long)i * NFILT,
        w2_sw[i], ue_b2 + (long)i * NFILT, tbuf, e3row, e3col, agg);
    gemm_f32(stream, agg, uv1_sw[i], uv1_b + (long)i * HIDDIM, nb1,
             NNODES, HIDDIM, NFILT, NFILT, HIDDIM, 2, 0);
    gemm_f32(stream, nb1, uv2_sw[i], uv2_b + (long)i * HIDDIM, vbuf,
             NNODES, HIDDIM, HIDDIM, HIDDIM, HIDDIM, 0, 1);   // v += upd
  }

  // 4) GINE stack
  for (int i = 0; i < NLAYER; ++i) {
    zero(stream, nb2, NH);  // agg2
    gine_edge_kernel<<<NE2 / (16 * 8), 256, 0, stream>>>(
        epad, el_sw[i], el_b + (long)i * HIDDIM, hbuf, e2src, e2dst, nb2);
    add_kernel<<<(int)(NH / 256), 256, 0, stream>>>(hbuf, nb2, nb1, NH);
    gemm_f32(stream, nb1, nn_sw, nn_b, hbuf, NNODES, HIDDIM, HIDDIM,
             HIDDIM, HIDDIM, 1, 0);          // relu(relu(.)) == relu(.)
  }

  // 5) pools + branches
  zero(stream, gnn_p, (long)NGRAPH * HIDDIM);
  pool_kernel<<<(int)(NH / 256), 256, 0, stream>>>(hbuf, batch, gnn_p);

  gemm_f32(stream, fp, fpn1_sw, fpn_b1, gt1, NGRAPH, HIDDIM, FPSZ, FPSZ, HIDDIM, 1, 0);
  gemm_f32(stream, gt1, fpn2_sw, fpn_b2, fcn, NGRAPH, HIDDIM, HIDDIM, HIDDIM, HIDDIM, 0, 0);

  gemm_f32(stream, vbuf, uu1_sw, uu_b1, nb1, NNODES, HIDDIM / 2, HIDDIM,
           HIDDIM, HIDDIM / 2, 2, 0);
  gemm_f32(stream, nb1, uu2_sw, uu_b2, nb2, NNODES, HIDDIM, HIDDIM / 2,
           HIDDIM / 2, HIDDIM, 0, 0);
  zero(stream, sch_p, (long)NGRAPH * HIDDIM);
  pool_kernel<<<(int)(NH / 256), 256, 0, stream>>>(nb2, batch, sch_p);

  // 6) final readout
  cat3_kernel<<<(NGRAPH * 3 * HIDDIM + 255) / 256, 256, 0, stream>>>(gnn_p, fcn, sch_p, cat3);
  gemm_f32(stream, cat3, hidw_sw, hid_b, gt2, NGRAPH, HIDDIM, 3 * HIDDIM,
           3 * HIDDIM, HIDDIM, 1, 0);
  gemm_f32(stream, gt2, outw_sw, out_b, out_p, NGRAPH, NTASK, HIDDIM,
           HIDDIM, NTASK, 0, 0);
}
